// SparseLinear_44135083933804
// MI455X (gfx1250) — matrix operand order repaired and checked by reading, not compiled, not run on hardware
//
#include <hip/hip_runtime.h>

#define IN_F   8192
#define OUT_F  8192
#define BATCH  1024
#define NNZ_N  671088

#define BM 128
#define BN 128
#define BK 32
#define LDSW 40                      // padded LDS row stride (80 B) = 32 elems + 4-dword pad
#define TILE_BYTES (BM * LDSW * 2)   // 10240 B per tile
#define BUF_BYTES  (2 * TILE_BYTES)  // A+B per buffer
#define SMEM_BYTES (2 * BUF_BYTES)   // double buffered: 40960 B

typedef __attribute__((ext_vector_type(16))) __bf16 v16bf;
typedef __attribute__((ext_vector_type(8)))  float  v8f;
typedef __attribute__((ext_vector_type(4)))  unsigned int v4u;
typedef __attribute__((ext_vector_type(8)))  int v8i;
typedef __attribute__((ext_vector_type(4)))  int v4i;

__device__ __forceinline__ unsigned short f2bf(float f) {
  union { float f; unsigned u; } c; c.f = f;
  unsigned u = c.u + 0x7FFFu + ((c.u >> 16) & 1u);   // round-to-nearest-even
  return (unsigned short)(u >> 16);
}

// TDM: DMA a 2D tile (tile_d0 x tile_d1, 2-byte elems) from global into LDS,
// padding each 16-dword row with 4 dwords so LDS row stride = 80 B (LDSW).
__device__ __forceinline__ void tdm_load_2d(const void* gptr, unsigned lds_off,
                                            unsigned tensor_d0, unsigned tensor_d1,
                                            unsigned tile_d0, unsigned tile_d1,
                                            unsigned long long stride_e) {
  unsigned long long ga = (unsigned long long)gptr;
  v4u g0;
  g0[0] = 1u;                                   // count=1, user descriptor
  g0[1] = lds_off;                              // LDS byte address
  g0[2] = (unsigned)ga;                         // global_addr[31:0]
  g0[3] = (unsigned)((ga >> 32) & 0x01FFFFFFu)  // global_addr[56:32]
        | (2u << 30);                           // type=2 ("image")
  v8i g1;
  g1[0] = (int)((1u << 16)      // data_size = 1 -> 2-byte elements
              | (1u << 20)      // pad_enable
              | (3u << 22)      // pad_interval: 16 dwords (one 64-B row)
              | (3u << 25));    // pad_amount: 4 dwords (16 B)
  g1[1] = (int)((tensor_d0 & 0xFFFFu) << 16);                  // tensor_dim0[15:0]
  g1[2] = (int)((tensor_d0 >> 16) | ((tensor_d1 & 0xFFFFu) << 16));
  g1[3] = (int)((tensor_d1 >> 16) | (tile_d0 << 16));          // tile_dim0
  g1[4] = (int)tile_d1;                                        // tile_dim1 (tile_dim2=0)
  g1[5] = (int)(unsigned)(stride_e & 0xFFFFFFFFu);             // dim0_stride[31:0]
  g1[6] = (int)(unsigned)((stride_e >> 32) & 0xFFFFu);         // dim0_stride[47:32]
  g1[7] = 0;
  v4i g2 = {0, 0, 0, 0};
  v4i g3 = {0, 0, 0, 0};
  v8i g4 = {0, 0, 0, 0, 0, 0, 0, 0};  // extra group (6-arg toolchain form), unused
  __builtin_amdgcn_tensor_load_to_lds(g0, g1, g2, g3, g4, 0);
}

__global__ void zero_f32_kernel(float4* __restrict__ p, long n4) {
  long i = (long)blockIdx.x * blockDim.x + threadIdx.x;
  long stride = (long)gridDim.x * blockDim.x;
  float4 z = make_float4(0.f, 0.f, 0.f, 0.f);
  for (; i < n4; i += stride) p[i] = z;
}

// Scatter COO values into W^T (OUT_F x IN_F, row-major): WT[col][row] += v
__global__ void scatter_T_kernel(const float* __restrict__ vals,
                                 const int* __restrict__ rows,
                                 const int* __restrict__ cols,
                                 float* __restrict__ WT, int nnz) {
  int i = blockIdx.x * blockDim.x + threadIdx.x;
  if (i < nnz) {
    long idx = (long)cols[i] * IN_F + (long)rows[i];
    atomicAdd(&WT[idx], vals[i]);
  }
}

__global__ void cvt_bf16_kernel(const float* __restrict__ src,
                                unsigned short* __restrict__ dst, long n) {
  long i = ((long)blockIdx.x * blockDim.x + threadIdx.x) * 4;
  long stride = (long)gridDim.x * blockDim.x * 4;
  for (; i < n; i += stride) {
    float4 f = *(const float4*)(src + i);
    union { unsigned short s[4]; uint2 u; } o;
    o.s[0] = f2bf(f.x); o.s[1] = f2bf(f.y); o.s[2] = f2bf(f.z); o.s[3] = f2bf(f.w);
    *(uint2*)(dst + i) = o.u;
  }
}

// C[M,N] = Xbf[M,K] * WbfT[N,K]^T + bias; TDM-fed double-buffered LDS, bf16 WMMA.
__global__ __launch_bounds__(256)
void gemm_wmma_kernel(const unsigned short* __restrict__ Xbf,
                      const unsigned short* __restrict__ WT,
                      const float* __restrict__ bias,
                      float* __restrict__ out) {
  extern __shared__ unsigned short smem[];   // [buf0: A,B][buf1: A,B]

  const int tid  = threadIdx.x;
  const int lane = tid & 31;
  const int wave = tid >> 5;
  const int wm   = wave & 1;   // 64-row slab
  const int wn   = wave >> 1;  // 32-col slab
  const int l16  = lane & 15;
  const int g    = lane >> 4;

  const int m0 = blockIdx.y * BM;
  const int n0 = blockIdx.x * BN;

  const unsigned short* Arow = Xbf + (size_t)m0 * IN_F;   // A tile row base
  const unsigned short* Brow = WT  + (size_t)n0 * IN_F;   // B tile row base

  v8f acc[4][2] = {};

  // Prologue: DMA tile k=0 into buffer 0.
  if (wave == 0)
    tdm_load_2d(Arow, 0u, IN_F, BATCH, BK, BM, IN_F);
  else if (wave == 1)
    tdm_load_2d(Brow, (unsigned)TILE_BYTES, IN_F, OUT_F, BK, BN, IN_F);
  if (wave < 2) __builtin_amdgcn_s_wait_tensorcnt(0);
  __syncthreads();

  unsigned off = 0;   // byte offset of current buffer (0 or BUF_BYTES)
  for (int k0 = 0; k0 < IN_F; k0 += BK) {
    const unsigned nxt = off ^ (unsigned)BUF_BYTES;
    // Kick off DMA of the next K tile while this one is consumed.
    if (k0 + BK < IN_F) {
      if (wave == 0)
        tdm_load_2d(Arow + (k0 + BK), nxt, IN_F, BATCH, BK, BM, IN_F);
      else if (wave == 1)
        tdm_load_2d(Brow + (k0 + BK), nxt + (unsigned)TILE_BYTES, IN_F, OUT_F, BK, BN, IN_F);
    }

    const unsigned short* Al = smem + (off >> 1);
    const unsigned short* Bl = smem + ((off + (unsigned)TILE_BYTES) >> 1);

    union FragU { uint4 u[2]; v16bf v; };
    FragU af[4], bfr[2];
    // A frag (16x32): lane = row M=l16; elems 0..7 = K[8g..8g+7], 8..15 = K[16+8g..+7]
    #pragma unroll
    for (int sm = 0; sm < 4; ++sm) {
      const unsigned short* p = Al + (wm * 64 + sm * 16 + l16) * LDSW;
      af[sm].u[0] = *(const uint4*)(p + g * 8);
      af[sm].u[1] = *(const uint4*)(p + 16 + g * 8);
    }
    // B frag (32x16): lane = col N=l16; elems 0..7 = K[8g..8g+7], 8..15 = K[16+8g..+7]
    #pragma unroll
    for (int sn = 0; sn < 2; ++sn) {
      const unsigned short* p = Bl + (wn * 32 + sn * 16 + l16) * LDSW;
      bfr[sn].u[0] = *(const uint4*)(p + g * 8);
      bfr[sn].u[1] = *(const uint4*)(p + 16 + g * 8);
    }
    #pragma unroll
    for (int sm = 0; sm < 4; ++sm)
      #pragma unroll
      for (int sn = 0; sn < 2; ++sn)
        acc[sm][sn] = __builtin_amdgcn_wmma_f32_16x16x32_bf16(
            false, af[sm].v, false, bfr[sn].v,
            (short)0, acc[sm][sn], false, false);

    // Single rendezvous: issuing waves prove DMA[nxt] complete before signaling;
    // barrier completion also proves all waves finished reading buf[off].
    if (wave < 2) __builtin_amdgcn_s_wait_tensorcnt(0);
    __syncthreads();
    off = nxt;
  }

  // Epilogue: C/D layout: VGPR v, lane l -> row = 8*(l>>4)+v, col = l&15
  #pragma unroll
  for (int sn = 0; sn < 2; ++sn) {
    const int col = n0 + wn * 32 + sn * 16 + l16;
    const float bv = bias[col];
    #pragma unroll
    for (int sm = 0; sm < 4; ++sm) {
      const int rbase = m0 + wm * 64 + sm * 16 + g * 8;
      #pragma unroll
      for (int v = 0; v < 8; ++v)
        out[(long)(rbase + v) * OUT_F + col] = acc[sm][sn][v] + bv;
    }
  }
}

extern "C" void kernel_launch(void* const* d_in, const int* in_sizes, int n_in,
                              void* d_out, int out_size, void* d_ws, size_t ws_size,
                              hipStream_t stream) {
  const float* x      = (const float*)d_in[0];
  const float* values = (const float*)d_in[1];
  const float* bias   = (const float*)d_in[2];
  const int*   rows   = (const int*)d_in[3];
  const int*   cols   = (const int*)d_in[4];
  float* out = (float*)d_out;

  const size_t nW = (size_t)IN_F * OUT_F;
  float*          W32 = (float*)d_ws;                                     // 256 MB
  unsigned short* WTb = (unsigned short*)((char*)d_ws + nW * 4);          // 128 MB
  unsigned short* Xb  = (unsigned short*)((char*)d_ws + nW * 4 + nW * 2); // 16 MB

  zero_f32_kernel<<<8192, 256, 0, stream>>>((float4*)W32, (long)(nW / 4));
  scatter_T_kernel<<<(NNZ_N + 255) / 256, 256, 0, stream>>>(values, rows, cols, W32, NNZ_N);
  cvt_bf16_kernel<<<8192, 256, 0, stream>>>(W32, WTb, (long)nW);
  cvt_bf16_kernel<<<2048, 256, 0, stream>>>(x, Xb, (long)BATCH * IN_F);

  dim3 grid(OUT_F / BN, BATCH / BM);
  gemm_wmma_kernel<<<grid, 256, SMEM_BYTES, stream>>>(Xb, WTb, bias, out);
}
